// ExpertParallelMoE_18451179504164
// MI455X (gfx1250) — compile-verified, matrix operand order checked by reading, not run on hardware
//
#include <hip/hip_runtime.h>
#include <hip/hip_bf16.h>
#include <math.h>

// ---------------------------------------------------------------------------
// MoE (8 experts, top-2) SwiGLU FFN for MI455X (gfx1250, wave32, WMMA).
//
// Routed grouped-GEMM pipeline (103 GFLOP vs the reference's dense 412 GFLOP),
// bf16 WMMA (v_wmma_f32_16x16x32_bf16) with fp32 accumulation. bf16 weight
// copies (~100 MB) stay resident in the 192 MB L2. Each wave register-blocks
// a 16x32 output tile (2.5-3 b128 loads per WMMA), and the K-loop is
// explicitly double-buffered so next-iteration fragment loads are in flight
// while the current WMMAs execute (partial s_wait_loadcnt, no full drains).
// ---------------------------------------------------------------------------

#define NEXP  8
#define DM    2048          // d_model
#define DFF   1024          // d_ff
#define NTOK  4096          // BATCH*SEQ
#define NPAIR (2*NTOK)      // token-expert pairs (top-2)

typedef __bf16 bf16_t;
typedef __attribute__((ext_vector_type(16))) __bf16 v16bf;
typedef __attribute__((ext_vector_type(8)))  __bf16 v8bf;
typedef __attribute__((ext_vector_type(8)))  float  v8f;

struct PairEntry { int tok_k; float w; };   // tok_k = token*2 + k

// --- WMMA helpers -----------------------------------------------------------

// Per-lane 16-bit fragment (A row / B column) for 16x16x32:
//   lanes 0-15 : elems 0..7 = K0..7,  elems 8..15 = K16..23
//   lanes 16-31: elems 0..7 = K8..15, elems 8..15 = K24..31
__device__ __forceinline__ v16bf ld_frag(const bf16_t* p, int klo) {
    v8bf a = *(const v8bf*)(p + klo);        // 16B -> global_load_b128
    v8bf b = *(const v8bf*)(p + klo + 16);
    v16bf r;
#pragma unroll
    for (int i = 0; i < 8; ++i) { r[i] = a[i]; r[i + 8] = b[i]; }
    return r;
}

__device__ __forceinline__ v8f wmma_bf16(v16bf a, v16bf b, v8f c) {
    return __builtin_amdgcn_wmma_f32_16x16x32_bf16(
        /*neg_a=*/false, a, /*neg_b=*/false, b,
        /*c_mod=*/(short)0, c, /*reuse_a=*/false, /*reuse_b=*/false);
}

// --- init -------------------------------------------------------------------

__global__ void zero_kernel(float* __restrict__ out, int n, int* __restrict__ counts) {
    int i = blockIdx.x * blockDim.x + threadIdx.x;
    if (i < n) out[i] = 0.0f;
    if (i < NEXP) counts[i] = 0;
}

__global__ void cvt_bf16_kernel(const float* __restrict__ in, bf16_t* __restrict__ out, int n) {
    int i = blockIdx.x * blockDim.x + threadIdx.x;
    if (i < n) out[i] = (bf16_t)in[i];
}

// fp32 [R][C] -> bf16 [C][R] per expert (LDS tiled; R,C multiples of 32 here)
__global__ void transpose_cvt_kernel(const float* __restrict__ in, bf16_t* __restrict__ out,
                                     int R, int C) {
    __shared__ bf16_t tile[32][33];
    const int e = blockIdx.z;
    const float* I = in + (size_t)e * R * C;
    bf16_t* O = out + (size_t)e * R * C;
    const int c0 = blockIdx.x * 32, r0 = blockIdx.y * 32;
#pragma unroll
    for (int i = threadIdx.y; i < 32; i += 8) {
        tile[i][threadIdx.x] = (bf16_t)I[(size_t)(r0 + i) * C + (c0 + threadIdx.x)];
    }
    __syncthreads();
#pragma unroll
    for (int i = threadIdx.y; i < 32; i += 8) {
        O[(size_t)(c0 + i) * R + (r0 + threadIdx.x)] = tile[threadIdx.x][i];
    }
}

// --- router: logits, top-2, softmax, scatter into per-expert lists ----------

__global__ void router_kernel(const float* __restrict__ x,      // [NTOK][DM]
                              const float* __restrict__ rw,     // [NEXP][DM]
                              int* __restrict__ counts,         // [NEXP]
                              PairEntry* __restrict__ lists) {  // [NEXP][NTOK]
    const int n = blockIdx.x;
    const float* xr = x + (size_t)n * DM;
    float acc[NEXP] = {};
    for (int d = threadIdx.x; d < DM; d += 256) {
        const float xv = xr[d];
#pragma unroll
        for (int e = 0; e < NEXP; ++e) acc[e] += xv * rw[e * DM + d];
    }
    __shared__ float s[NEXP][256];
#pragma unroll
    for (int e = 0; e < NEXP; ++e) s[e][threadIdx.x] = acc[e];
    __syncthreads();
    for (int off = 128; off > 0; off >>= 1) {
        if (threadIdx.x < off) {
#pragma unroll
            for (int e = 0; e < NEXP; ++e) s[e][threadIdx.x] += s[e][threadIdx.x + off];
        }
        __syncthreads();
    }
    if (threadIdx.x == 0) {
        int e0 = 0; float v0 = s[0][0];
#pragma unroll
        for (int e = 1; e < NEXP; ++e) if (s[e][0] > v0) { v0 = s[e][0]; e0 = e; }
        int e1 = (e0 == 0) ? 1 : 0; float v1 = s[e1][0];
#pragma unroll
        for (int e = 0; e < NEXP; ++e)
            if (e != e0 && s[e][0] > v1) { v1 = s[e][0]; e1 = e; }
        const float ex = __expf(v1 - v0);         // softmax over {v0, v1}
        const float inv = 1.0f / (1.0f + ex);
        const float w0 = inv, w1 = ex * inv;
        int s0 = atomicAdd(&counts[e0], 1);
        lists[e0 * NTOK + s0] = PairEntry{2 * n + 0, w0};
        int s1 = atomicAdd(&counts[e1], 1);
        lists[e1 * NTOK + s1] = PairEntry{2 * n + 1, w1};
    }
}

// --- grouped GEMM 1: h = silu(x@w1) * (x@w2), gathered per expert -----------
// block: 256 threads = 8 waves; block tile = 16 tokens x 256 F-cols.
// wave tile = 16x32 (cols c0 and c0+16): one A fragment feeds 4 WMMAs.
// Explicit double-buffering over K: loads for step k+1 issue before the
// WMMAs of step k.

__global__ void __launch_bounds__(256)
moe_gemm1_kernel(const bf16_t* __restrict__ xb,   // [NTOK][DM]
                 const bf16_t* __restrict__ w1t,  // [E][DFF][DM]
                 const bf16_t* __restrict__ w2t,  // [E][DFF][DM]
                 const int* __restrict__ counts,
                 const PairEntry* __restrict__ lists,
                 bf16_t* __restrict__ h) {        // [NPAIR][DFF]
    const int e = blockIdx.z;
    const int cnt = counts[e];
    const int mtile = blockIdx.y;
    if (mtile * 16 >= cnt) return;

    const int lane = threadIdx.x & 31;
    const int wave = threadIdx.x >> 5;
    const int col0 = blockIdx.x * 256 + wave * 32 + (lane & 15);
    const int col1 = col0 + 16;

    __shared__ int s_tok[16];
    if (threadIdx.x < 16) {
        int slot = mtile * 16 + threadIdx.x;
        if (slot >= cnt) slot = cnt - 1;               // clamp (stores masked)
        s_tok[threadIdx.x] = lists[e * NTOK + slot].tok_k;
    }
    __syncthreads();

    const bf16_t* arow = xb + (size_t)(s_tok[lane & 15] >> 1) * DM;
    const bf16_t* b1r0 = w1t + ((size_t)e * DFF + col0) * DM;
    const bf16_t* b1r1 = w1t + ((size_t)e * DFF + col1) * DM;
    const bf16_t* b2r0 = w2t + ((size_t)e * DFF + col0) * DM;
    const bf16_t* b2r1 = w2t + ((size_t)e * DFF + col1) * DM;
    const int klo = (lane & 16) ? 8 : 0;

    v8f acc10 = {}, acc11 = {}, acc20 = {}, acc21 = {};

    // prologue: fragments for k = 0
    v16bf a   = ld_frag(arow, klo);
    v16bf b10 = ld_frag(b1r0, klo);
    v16bf b11 = ld_frag(b1r1, klo);
    v16bf b20 = ld_frag(b2r0, klo);
    v16bf b21 = ld_frag(b2r1, klo);

    for (int k = 32; k < DM; k += 32) {
        __builtin_prefetch(b1r0 + k + 256, 0, 1);      // global_prefetch_b8
        __builtin_prefetch(b2r0 + k + 256, 0, 1);
        // issue next-step loads first so they overlap this step's WMMAs
        v16bf an   = ld_frag(arow + k, klo);
        v16bf b10n = ld_frag(b1r0 + k, klo);
        v16bf b11n = ld_frag(b1r1 + k, klo);
        v16bf b20n = ld_frag(b2r0 + k, klo);
        v16bf b21n = ld_frag(b2r1 + k, klo);
        acc10 = wmma_bf16(a, b10, acc10);
        acc11 = wmma_bf16(a, b11, acc11);
        acc20 = wmma_bf16(a, b20, acc20);
        acc21 = wmma_bf16(a, b21, acc21);
        a = an; b10 = b10n; b11 = b11n; b20 = b20n; b21 = b21n;
    }
    // epilogue step
    acc10 = wmma_bf16(a, b10, acc10);
    acc11 = wmma_bf16(a, b11, acc11);
    acc20 = wmma_bf16(a, b20, acc20);
    acc21 = wmma_bf16(a, b21, acc21);

    const int mh = (lane >> 4) * 8;                    // C layout: row = v + 8*(lane>=16)
#pragma unroll
    for (int v = 0; v < 8; ++v) {
        const int m = v + mh;
        if (mtile * 16 + m < cnt) {
            bf16_t* hrow = h + (size_t)s_tok[m] * DFF;
            float g0 = acc10[v];
            float g1 = acc11[v];
            hrow[col0] = (bf16_t)((g0 / (1.0f + __expf(-g0))) * acc20[v]);  // silu*val
            hrow[col1] = (bf16_t)((g1 / (1.0f + __expf(-g1))) * acc21[v]);
        }
    }
}

// --- grouped GEMM 2: y = h @ w3, scale by routing weight, scatter-add -------
// block tile = 16 tokens x 256 D-cols; wave tile = 16x32 (2 accumulators).

__global__ void __launch_bounds__(256)
moe_gemm2_kernel(const bf16_t* __restrict__ h,    // [NPAIR][DFF]
                 const bf16_t* __restrict__ w3t,  // [E][DM][DFF]
                 const int* __restrict__ counts,
                 const PairEntry* __restrict__ lists,
                 float* __restrict__ out) {       // [NTOK][DM]
    const int e = blockIdx.z;
    const int cnt = counts[e];
    const int mtile = blockIdx.y;
    if (mtile * 16 >= cnt) return;

    const int lane = threadIdx.x & 31;
    const int wave = threadIdx.x >> 5;
    const int col0 = blockIdx.x * 256 + wave * 32 + (lane & 15);
    const int col1 = col0 + 16;

    __shared__ int   s_tok[16];
    __shared__ float s_w[16];
    if (threadIdx.x < 16) {
        int slot = mtile * 16 + threadIdx.x;
        if (slot >= cnt) slot = cnt - 1;
        PairEntry pe = lists[e * NTOK + slot];
        s_tok[threadIdx.x] = pe.tok_k;
        s_w[threadIdx.x] = pe.w;
    }
    __syncthreads();

    const bf16_t* arow = h + (size_t)s_tok[lane & 15] * DFF;
    const bf16_t* br0  = w3t + ((size_t)e * DM + col0) * DFF;
    const bf16_t* br1  = w3t + ((size_t)e * DM + col1) * DFF;
    const int klo = (lane & 16) ? 8 : 0;

    v8f acc0 = {}, acc1 = {};

    v16bf a  = ld_frag(arow, klo);
    v16bf b0 = ld_frag(br0, klo);
    v16bf b1 = ld_frag(br1, klo);

    for (int k = 32; k < DFF; k += 32) {
        __builtin_prefetch(br0 + k + 256, 0, 1);
        v16bf an  = ld_frag(arow + k, klo);
        v16bf b0n = ld_frag(br0 + k, klo);
        v16bf b1n = ld_frag(br1 + k, klo);
        acc0 = wmma_bf16(a, b0, acc0);
        acc1 = wmma_bf16(a, b1, acc1);
        a = an; b0 = b0n; b1 = b1n;
    }
    acc0 = wmma_bf16(a, b0, acc0);
    acc1 = wmma_bf16(a, b1, acc1);

    const int mh = (lane >> 4) * 8;
#pragma unroll
    for (int v = 0; v < 8; ++v) {
        const int m = v + mh;
        if (mtile * 16 + m < cnt) {
            float* orow = out + (size_t)(s_tok[m] >> 1) * DM;
            const float w = s_w[m];
            atomicAdd(&orow[col0], acc0[v] * w);   // exactly 2 commutative adds
            atomicAdd(&orow[col1], acc1[v] * w);   // per element -> deterministic
        }
    }
}

// ---------------------------------------------------------------------------

extern "C" void kernel_launch(void* const* d_in, const int* in_sizes, int n_in,
                              void* d_out, int out_size, void* d_ws, size_t ws_size,
                              hipStream_t stream) {
    (void)in_sizes; (void)n_in; (void)out_size; (void)ws_size;
    const float* x  = (const float*)d_in[0];   // [2,2048,2048]
    const float* rw = (const float*)d_in[1];   // [8,2048]
    const float* w1 = (const float*)d_in[2];   // [8,2048,1024]
    const float* w2 = (const float*)d_in[3];   // [8,2048,1024]
    const float* w3 = (const float*)d_in[4];   // [8,1024,2048]
    float* out = (float*)d_out;                // [2,2048,2048]

    // Workspace carve (~134.5 MB total)
    char* ws = (char*)d_ws;
    size_t off = 0;
    auto carve = [&](size_t bytes) -> void* {
        void* p = ws + off;
        off = (off + bytes + 255) & ~(size_t)255;
        return p;
    };
    bf16_t* xb     = (bf16_t*)carve((size_t)NTOK * DM * sizeof(bf16_t));        // 16.8 MB
    bf16_t* w1t    = (bf16_t*)carve((size_t)NEXP * DFF * DM * sizeof(bf16_t));  // 33.6 MB
    bf16_t* w2t    = (bf16_t*)carve((size_t)NEXP * DFF * DM * sizeof(bf16_t));  // 33.6 MB
    bf16_t* w3t    = (bf16_t*)carve((size_t)NEXP * DM * DFF * sizeof(bf16_t));  // 33.6 MB
    bf16_t* hbuf   = (bf16_t*)carve((size_t)NPAIR * DFF * sizeof(bf16_t));      // 16.8 MB
    PairEntry* lists = (PairEntry*)carve((size_t)NEXP * NTOK * sizeof(PairEntry));
    int* counts    = (int*)carve(NEXP * sizeof(int));

    const int nOut = NTOK * DM;

    // 1) zero output + counters
    zero_kernel<<<(nOut + 255) / 256, 256, 0, stream>>>(out, nOut, counts);

    // 2) precision conversion / weight transposition (bf16 copies live in L2)
    cvt_bf16_kernel<<<(nOut + 255) / 256, 256, 0, stream>>>(x, xb, nOut);
    {
        dim3 blk(32, 8);
        dim3 g12(DFF / 32, DM / 32, NEXP);   // [D][F] -> [F][D]
        transpose_cvt_kernel<<<g12, blk, 0, stream>>>(w1, w1t, DM, DFF);
        transpose_cvt_kernel<<<g12, blk, 0, stream>>>(w2, w2t, DM, DFF);
        dim3 g3(DM / 32, DFF / 32, NEXP);    // [F][D] -> [D][F]
        transpose_cvt_kernel<<<g3, blk, 0, stream>>>(w3, w3t, DFF, DM);
    }

    // 3) routing (top-2 + softmax + expert list scatter)
    router_kernel<<<NTOK, 256, 0, stream>>>(x, rw, counts, lists);

    // 4) grouped GEMM 1: fused x@w1 / x@w2 + SwiGLU  (WMMA bf16, 16x32/wave)
    {
        dim3 g(DFF / 256, NTOK / 16, NEXP);  // (4, 256, 8); early-exit past count
        moe_gemm1_kernel<<<g, 256, 0, stream>>>(xb, w1t, w2t, counts, lists, hbuf);
    }

    // 5) grouped GEMM 2: h@w3, weighted scatter-add into output
    {
        dim3 g(DM / 256, NTOK / 16, NEXP);   // (8, 256, 8)
        moe_gemm2_kernel<<<g, 256, 0, stream>>>(hbuf, w3t, counts, lists, out);
    }
}
